// FalconPerception_59261958750235
// MI455X (gfx1250) — compile-verified, matrix operand order checked by reading, not run on hardware
//
#include <hip/hip_runtime.h>
#include <stdint.h>

// ---------------------------------------------------------------------------
// Model constants (match reference)
// ---------------------------------------------------------------------------
#define Sq 1024
#define DIMq 2048
#define Hq 16
#define KVHq 4
#define HDq 128
#define HIDq 8192
#define Lq 2
#define VOCABq 32000
#define NIMGq 256
#define TOKSTARTq 1
#define IMGINq (2*14*14*3)   // 1176
#define EPSq 1e-6f

#define QDIMq (Hq*HDq)            // 2048
#define KVDIMq (KVHq*HDq)         // 512
#define QKVDq (QDIMq + 2*KVDIMq)  // 3072

typedef __bf16 bf16_t;
typedef __attribute__((ext_vector_type(4)))  float  v4f;
typedef __attribute__((ext_vector_type(8)))  float  v8f;
typedef __attribute__((ext_vector_type(16))) bf16_t v16bf;
typedef int v4i __attribute__((vector_size(16)));

// ---------------------------------------------------------------------------
// CDNA5 async global->LDS staging (GLOBAL_LOAD_ASYNC_TO_LDS_B128, tracked by
// ASYNCcnt). Builtin confirmed present; signature:
//   (AS1 int4* gaddr, AS3 int4* lds, imm offset, imm cpol)
// Guarded so a missing builtin still falls back to a sync register copy.
// ---------------------------------------------------------------------------
#if defined(__has_builtin)
#  if __has_builtin(__builtin_amdgcn_global_load_async_to_lds_b128) && \
      __has_builtin(__builtin_amdgcn_s_wait_asynccnt)
#    define ASYNC_OK 1
#  endif
#endif
#ifndef ASYNC_OK
#  define ASYNC_OK 0
#endif

__device__ __forceinline__ void cp16_g2l(const float* g, float* l)
{
#if ASYNC_OK
    // flat-LDS low 32 bits == LDS offset (ISA 10.2 aperture rules)
    __builtin_amdgcn_global_load_async_to_lds_b128(
        (__attribute__((address_space(1))) v4i*)(uintptr_t)g,
        (__attribute__((address_space(3))) v4i*)(unsigned int)(uintptr_t)l,
        0, 0);
#else
    *(v4f*)l = *(const v4f*)g;
#endif
}

__device__ __forceinline__ v16bf cvt16(v4f a, v4f b, v4f c, v4f d)
{
    v16bf r;
    r[0]=(bf16_t)a[0];  r[1]=(bf16_t)a[1];  r[2]=(bf16_t)a[2];  r[3]=(bf16_t)a[3];
    r[4]=(bf16_t)b[0];  r[5]=(bf16_t)b[1];  r[6]=(bf16_t)b[2];  r[7]=(bf16_t)b[3];
    r[8]=(bf16_t)c[0];  r[9]=(bf16_t)c[1];  r[10]=(bf16_t)c[2]; r[11]=(bf16_t)c[3];
    r[12]=(bf16_t)d[0]; r[13]=(bf16_t)d[1]; r[14]=(bf16_t)d[2]; r[15]=(bf16_t)d[3];
    return r;
}

// ---------------------------------------------------------------------------
// bf16-WMMA GEMM:  C[M,N] (f32)  (+)=  A[M,K] (f32) * W[N,K]^T (f32)
// Block: 256 threads = 8 wave32. Macro-tile BM=64 x BN=128, BK=32.
// Each wave owns a 16x64 strip: 1 A fragment x 4 B fragments -> 4 WMMAs/step.
// Double-buffered LDS (f32), async global->LDS staging when available.
// EDGE=1 only for the K-unaligned image projection GEMM.
// grid = (N/128, M/64); all launches are exact multiples.
// ---------------------------------------------------------------------------
#define BM 64
#define BN 128
#define BK 32
#define LDSF 36   // padded f32 row stride: 144 bytes -> every row 16B aligned

template<int EDGE>
__global__ __launch_bounds__(256)
void fp_gemm(const float* __restrict__ A,
             const float* __restrict__ W,
             float* __restrict__ C,
             int M, int N, int K, int ldc, int beta)
{
    __shared__ __align__(16) float As[2][BM][LDSF];
    __shared__ __align__(16) float Ws[2][BN][LDSF];

    const int tid  = threadIdx.x;
    const int lane = tid & 31;
    const int wid  = tid >> 5;            // 0..7
    const int half = lane >> 4;           // 0,1
    const int m16  = lane & 15;
    const int kbase = half * 8;           // K-group base per 16-bit WMMA layout

    const int bm = blockIdx.y * BM;
    const int bn = blockIdx.x * BN;
    const int lm  = (wid >> 1) * 16;      // wave's M tile: 0,16,32,48
    const int ln0 = (wid & 1) * 64;       // wave's N strip: 0 or 64

    const int nkt = (K + BK - 1) / BK;

    auto stage = [&](int kt, int buf) {
        const int k0 = kt * BK;
        // A tile: 64x32 f32 = 512 x 16B chunks, 2 per thread
        #pragma unroll
        for (int q = 0; q < 2; ++q) {
            const int chunk = q * 256 + tid;
            const int row = chunk >> 3;
            const int col = (chunk & 7) * 4;
            float* dst = &As[buf][row][col];
            if (EDGE) {
                const float* ap = A + (size_t)(bm + row) * K;
                const int kk = k0 + col;
                v4f v;
                if (kk + 3 < K) v = *(const v4f*)(ap + kk);
                else {
                    v[0] = (kk+0 < K) ? ap[kk+0] : 0.f;
                    v[1] = (kk+1 < K) ? ap[kk+1] : 0.f;
                    v[2] = (kk+2 < K) ? ap[kk+2] : 0.f;
                    v[3] = (kk+3 < K) ? ap[kk+3] : 0.f;
                }
                *(v4f*)dst = v;
            } else {
                cp16_g2l(A + (size_t)(bm + row) * K + k0 + col, dst);
            }
        }
        // W tile: 128x32 f32 = 1024 x 16B chunks, 4 per thread
        #pragma unroll
        for (int q = 0; q < 4; ++q) {
            const int chunk = q * 256 + tid;
            const int row = chunk >> 3;
            const int col = (chunk & 7) * 4;
            float* dst = &Ws[buf][row][col];
            if (EDGE) {
                const float* wp = W + (size_t)(bn + row) * K;
                const int kk = k0 + col;
                v4f v;
                if (kk + 3 < K) v = *(const v4f*)(wp + kk);
                else {
                    v[0] = (kk+0 < K) ? wp[kk+0] : 0.f;
                    v[1] = (kk+1 < K) ? wp[kk+1] : 0.f;
                    v[2] = (kk+2 < K) ? wp[kk+2] : 0.f;
                    v[3] = (kk+3 < K) ? wp[kk+3] : 0.f;
                }
                *(v4f*)dst = v;
            } else {
                cp16_g2l(W + (size_t)(bn + row) * K + k0 + col, dst);
            }
        }
    };

    v8f zero = {0.f,0.f,0.f,0.f,0.f,0.f,0.f,0.f};
    v8f acc[4] = {zero, zero, zero, zero};

    stage(0, 0);
    for (int kt = 0; kt < nkt; ++kt) {
        const int cur = kt & 1;
        if (kt + 1 < nkt) stage(kt + 1, cur ^ 1);
#if ASYNC_OK
        if (!EDGE) {
            // 6 async b128 ops issued per wave per stage; in-order completion
            if (kt + 1 < nkt) __builtin_amdgcn_s_wait_asynccnt(6);
            else              __builtin_amdgcn_s_wait_asynccnt(0);
        }
#endif
        __syncthreads();

        // A fragment (16-bit WMMA layout: lanes 0-15 -> K [0..7],[16..23];
        // lanes 16-31 -> K [8..15],[24..31]); convert f32->bf16 in-register.
        const float* arow = &As[cur][lm + m16][0];
        const v16bf afrag = cvt16(*(const v4f*)(arow + kbase),
                                  *(const v4f*)(arow + kbase + 4),
                                  *(const v4f*)(arow + kbase + 16),
                                  *(const v4f*)(arow + kbase + 20));
        #pragma unroll
        for (int j = 0; j < 4; ++j) {
            const float* brow = &Ws[cur][ln0 + j * 16 + m16][0];
            const v16bf bfrag = cvt16(*(const v4f*)(brow + kbase),
                                      *(const v4f*)(brow + kbase + 4),
                                      *(const v4f*)(brow + kbase + 16),
                                      *(const v4f*)(brow + kbase + 20));
            acc[j] = __builtin_amdgcn_wmma_f32_16x16x32_bf16(
                         false, afrag, false, bfrag, (short)0, acc[j], false, false);
        }
        __syncthreads();
    }

    // writeback; C/D layout: VGPR r -> M = r + 8*half, N = lane&15
    if (beta) {
        #pragma unroll
        for (int j = 0; j < 4; ++j) {
            const int col = bn + ln0 + j * 16 + m16;
            #pragma unroll
            for (int r = 0; r < 8; ++r) {
                const int row = bm + lm + half * 8 + r;
                C[(size_t)row * ldc + col] += acc[j][r];
            }
        }
    } else {
        #pragma unroll
        for (int j = 0; j < 4; ++j) {
            const int col = bn + ln0 + j * 16 + m16;
            #pragma unroll
            for (int r = 0; r < 8; ++r) {
                const int row = bm + lm + half * 8 + r;
                C[(size_t)row * ldc + col] = acc[j][r];
            }
        }
    }
}

// ---------------------------------------------------------------------------
// Token embedding gather
// ---------------------------------------------------------------------------
__global__ void fp_embed(const int* __restrict__ tokens,
                         const float* __restrict__ tok_emb,
                         float* __restrict__ x)
{
    const int s = blockIdx.x;
    const int tok = tokens[s];
    for (int d = threadIdx.x; d < DIMq; d += blockDim.x)
        x[(size_t)s * DIMq + d] = tok_emb[(size_t)tok * DIMq + d];
}

// ---------------------------------------------------------------------------
// RMSNorm over DIM=2048, optional weight
// ---------------------------------------------------------------------------
__global__ __launch_bounds__(256)
void fp_rms(const float* __restrict__ in, float* __restrict__ out,
            const float* __restrict__ w)
{
    __shared__ float red[256];
    const int s = blockIdx.x;
    const int t = threadIdx.x;
    const float* row = in + (size_t)s * DIMq;
    float ss = 0.f;
    for (int d = t; d < DIMq; d += 256) { float v = row[d]; ss += v * v; }
    red[t] = ss; __syncthreads();
    for (int o = 128; o > 0; o >>= 1) {
        if (t < o) red[t] += red[t + o];
        __syncthreads();
    }
    const float r = rsqrtf(red[0] / (float)DIMq + EPSq);
    float* orow = out + (size_t)s * DIMq;
    for (int d = t; d < DIMq; d += 256)
        orow[d] = row[d] * r * (w ? w[d] : 1.f);
}

// ---------------------------------------------------------------------------
// Fused per-head RMS + RoPE on q (16 heads) and k (4 heads) inside qkv buffer
// grid = (S, H+KVH), block = 128
// ---------------------------------------------------------------------------
__global__ __launch_bounds__(128)
void fp_qk_rms_rope(float* __restrict__ qkv,
                    const float* __restrict__ f1,   // (S, 32)
                    const float* __restrict__ f2)   // (S, 32)
{
    __shared__ float red[128];
    __shared__ float sh[HDq];
    const int s  = blockIdx.x;
    const int hh = blockIdx.y;                 // 0..19
    const int t  = threadIdx.x;
    const int off = (hh < Hq) ? hh * HDq : QDIMq + (hh - Hq) * HDq;
    float* row = qkv + (size_t)s * QKVDq + off;

    float v = row[t];
    red[t] = v * v; __syncthreads();
    for (int o = 64; o > 0; o >>= 1) {
        if (t < o) red[t] += red[t + o];
        __syncthreads();
    }
    const float r = rsqrtf(red[0] / (float)HDq + EPSq);
    sh[t] = v * r;
    __syncthreads();

    if (t < HDq / 2) {
        const int i = t;
        const float ang = (i < 32) ? f1[(size_t)s * 32 + i]
                                   : f2[(size_t)s * 32 + (i - 32)];
        const float c = cosf(ang), sn = sinf(ang);
        const float x1 = sh[2 * i], x2 = sh[2 * i + 1];
        row[2 * i]     = x1 * c  - x2 * sn;
        row[2 * i + 1] = x1 * sn + x2 * c;
    }
}

// ---------------------------------------------------------------------------
// Causal attention with LSE + sink scaling. grid = (S, H), block = 128.
// ---------------------------------------------------------------------------
__global__ __launch_bounds__(128)
void fp_attention(const float* __restrict__ qkv,
                  float* __restrict__ attn_out,
                  const float* __restrict__ sinks)   // (H,)
{
    __shared__ float shq[HDq];
    __shared__ float sc[Sq];
    __shared__ float red[128];

    const int s = blockIdx.x;
    const int h = blockIdx.y;
    const int t = threadIdx.x;
    const int kvh = h / (Hq / KVHq);
    const int nk = s + 1;
    const float scale = 0.08838834764831845f;   // 1/sqrt(128)

    const float* qrow  = qkv + (size_t)s * QKVDq + h * HDq;
    const float* kbase = qkv + QDIMq + (size_t)kvh * HDq;
    const float* vbase = qkv + QDIMq + KVDIMq + (size_t)kvh * HDq;

    shq[t] = qrow[t];
    __syncthreads();

    float lmax = -3.4e38f;
    for (int k = t; k < nk; k += 128) {
        const float* kr = kbase + (size_t)k * QKVDq;
        float d = 0.f;
        #pragma unroll 8
        for (int j = 0; j < HDq; ++j) d += shq[j] * kr[j];
        d *= scale;
        sc[k] = d;
        lmax = fmaxf(lmax, d);
    }
    red[t] = lmax; __syncthreads();
    for (int o = 64; o > 0; o >>= 1) {
        if (t < o) red[t] = fmaxf(red[t], red[t + o]);
        __syncthreads();
    }
    const float m = red[0];
    __syncthreads();

    float lsum = 0.f;
    for (int k = t; k < nk; k += 128) {
        const float p = __expf(sc[k] - m);
        sc[k] = p;
        lsum += p;
    }
    red[t] = lsum; __syncthreads();
    for (int o = 64; o > 0; o >>= 1) {
        if (t < o) red[t] += red[t + o];
        __syncthreads();
    }
    const float sum = red[0];
    const float lse = m + __logf(sum);
    const float inv = 1.f / sum;
    __syncthreads();

    for (int k = t; k < nk; k += 128) sc[k] *= inv;
    __syncthreads();

    float acc = 0.f;
    for (int k = 0; k < nk; ++k)
        acc += sc[k] * vbase[(size_t)k * QKVDq + t];

    const float sig = 1.f / (1.f + __expf(-(lse - sinks[h])));
    attn_out[(size_t)s * QDIMq + h * HDq + t] = acc * sig;
}

// ---------------------------------------------------------------------------
// squared-ReLU GLU: out[s,j] = relu(packed[s,2j])^2 * packed[s,2j+1]
// ---------------------------------------------------------------------------
__global__ void fp_glu(const float* __restrict__ packed,
                       float* __restrict__ out)
{
    const long long idx = (long long)blockIdx.x * blockDim.x + threadIdx.x;
    if (idx >= (long long)Sq * HIDq) return;
    const int s = (int)(idx / HIDq);
    const int j = (int)(idx % HIDq);
    const float g = packed[(size_t)s * (2 * HIDq) + 2 * j];
    const float u = packed[(size_t)s * (2 * HIDq) + 2 * j + 1];
    const float rg = fmaxf(g, 0.f);
    out[idx] = rg * rg * u;
}

// ---------------------------------------------------------------------------
// Host-side orchestration
// ---------------------------------------------------------------------------
static inline void launch_gemm(const float* A, const float* W, float* C,
                               int M, int N, int K, int ldc, int beta,
                               hipStream_t stream)
{
    dim3 grid((N + BN - 1) / BN, (M + BM - 1) / BM);
    if (K % BK == 0)
        fp_gemm<0><<<grid, 256, 0, stream>>>(A, W, C, M, N, K, ldc, beta);
    else
        fp_gemm<1><<<grid, 256, 0, stream>>>(A, W, C, M, N, K, ldc, beta);
}

extern "C" void kernel_launch(void* const* d_in, const int* in_sizes, int n_in,
                              void* d_out, int out_size, void* d_ws, size_t ws_size,
                              hipStream_t stream)
{
    (void)in_sizes; (void)n_in; (void)out_size; (void)ws_size;

    const int*   tokens    = (const int*)  d_in[0];
    const float* img_feats = (const float*)d_in[1];
    const float* freqs     = (const float*)d_in[2];
    const float* freqs2d   = (const float*)d_in[3];
    const float* tok_emb   = (const float*)d_in[4];
    const float* img_projw = (const float*)d_in[5];
    const float* wqkv      = (const float*)d_in[6];
    const float* wo        = (const float*)d_in[7];
    const float* sinks     = (const float*)d_in[8];
    const float* w13       = (const float*)d_in[9];
    const float* w2        = (const float*)d_in[10];
    const float* norm_w    = (const float*)d_in[11];
    const float* output_w  = (const float*)d_in[12];
    float* out = (float*)d_out;

    // workspace carve-up (all f32)
    char* ws = (char*)d_ws;
    float* x      = (float*)ws;  ws += (size_t)Sq * DIMq     * 4;
    float* h      = (float*)ws;  ws += (size_t)Sq * DIMq     * 4;
    float* qkv    = (float*)ws;  ws += (size_t)Sq * QKVDq    * 4;
    float* attn   = (float*)ws;  ws += (size_t)Sq * QDIMq    * 4;
    float* packed = (float*)ws;  ws += (size_t)Sq * 2 * HIDq * 4;
    float* glu    = (float*)ws;  ws += (size_t)Sq * HIDq     * 4;

    // 1) token embedding + image projection overwrite
    fp_embed<<<Sq, 256, 0, stream>>>(tokens, tok_emb, x);
    launch_gemm(img_feats, img_projw, x + (size_t)TOKSTARTq * DIMq,
                NIMGq, DIMq, IMGINq, DIMq, /*beta=*/0, stream);

    // 2) transformer layers
    for (int l = 0; l < Lq; ++l) {
        const float* wqkv_l = wqkv + (size_t)l * QKVDq * DIMq;
        const float* wo_l   = wo   + (size_t)l * DIMq * QDIMq;
        const float* w13_l  = w13  + (size_t)l * 2 * HIDq * DIMq;
        const float* w2_l   = w2   + (size_t)l * DIMq * HIDq;
        const float* sink_l = sinks + (size_t)l * Hq;

        fp_rms<<<Sq, 256, 0, stream>>>(x, h, nullptr);
        launch_gemm(h, wqkv_l, qkv, Sq, QKVDq, DIMq, QKVDq, 0, stream);

        fp_qk_rms_rope<<<dim3(Sq, Hq + KVHq), 128, 0, stream>>>(qkv, freqs, freqs2d);
        fp_attention<<<dim3(Sq, Hq), 128, 0, stream>>>(qkv, attn, sink_l);

        launch_gemm(attn, wo_l, x, Sq, DIMq, QDIMq, DIMq, /*beta=*/1, stream);

        fp_rms<<<Sq, 256, 0, stream>>>(x, h, nullptr);
        launch_gemm(h, w13_l, packed, Sq, 2 * HIDq, DIMq, 2 * HIDq, 0, stream);

        fp_glu<<<((long long)Sq * HIDq + 255) / 256, 256, 0, stream>>>(packed, glu);
        launch_gemm(glu, w2_l, x, Sq, DIMq, HIDq, DIMq, /*beta=*/1, stream);
    }

    // 3) final norm + vocab projection
    fp_rms<<<Sq, 256, 0, stream>>>(x, h, norm_w);
    launch_gemm(h, output_w, out, Sq, VOCABq, DIMq, VOCABq, 0, stream);
}
